// MessagePassing_47545287967105
// MI455X (gfx1250) — compile-verified, matrix operand order checked by reading, not run on hardware
//
#include <hip/hip_runtime.h>
#include <hip/hip_bf16.h>

// ---------------- problem constants ----------------
#define NN 50000
#define NE 800000
#define DD 64
#define UKDIM 192   // 2*64 + 64
#define MKDIM 128   // 2*64
#define TSTEPS 3

// padded LDS row strides (bf16 elements) -> conflict-free ds_load_b128
#define ULDS 200    // 192 + 8 pad : lane row offset r*36 mod 64 banks, stride-4 tiling
#define MLDS 136    // 128 + 8 pad : lane row offset r*4  mod 64 banks, stride-4 tiling

typedef __bf16 v16bf __attribute__((ext_vector_type(16)));
typedef float  v8f   __attribute__((ext_vector_type(8)));
typedef float  v4f   __attribute__((ext_vector_type(4)));

union ABfrag {
    v16bf v;
    __bf16 h[16];
    uint4  q[2];
};

__device__ __forceinline__ void cvt8_f32_bf16_nt(const float* __restrict__ p, __bf16* o) {
    v4f f0 = __builtin_nontemporal_load((const v4f*)(p));
    v4f f1 = __builtin_nontemporal_load((const v4f*)(p + 4));
    o[0] = (__bf16)f0[0]; o[1] = (__bf16)f0[1]; o[2] = (__bf16)f0[2]; o[3] = (__bf16)f0[3];
    o[4] = (__bf16)f1[0]; o[5] = (__bf16)f1[1]; o[6] = (__bf16)f1[2]; o[7] = (__bf16)f1[3];
}

__device__ __forceinline__ void cvt8_f32_bf16(const float* __restrict__ p, __bf16* o) {
    v4f f0 = *(const v4f*)(p);
    v4f f1 = *(const v4f*)(p + 4);
    o[0] = (__bf16)f0[0]; o[1] = (__bf16)f0[1]; o[2] = (__bf16)f0[2]; o[3] = (__bf16)f0[3];
    o[4] = (__bf16)f1[0]; o[5] = (__bf16)f1[1]; o[6] = (__bf16)f1[2]; o[7] = (__bf16)f1[3];
}

// ---------------- one-time prep ----------------
__global__ __launch_bounds__(256) void prep_weights_kernel(
    const float* __restrict__ U_w, const float* __restrict__ M_w,
    __bf16* __restrict__ Ubf, __bf16* __restrict__ Mbf)
{
    int i = blockIdx.x * blockDim.x + threadIdx.x;
    if (i < TSTEPS * DD * UKDIM) Ubf[i] = (__bf16)U_w[i];
    if (i < TSTEPS * DD * MKDIM) Mbf[i] = (__bf16)M_w[i];
}

__global__ __launch_bounds__(256) void init_kernel(
    const float* __restrict__ node_features, float* __restrict__ h, int* __restrict__ deg)
{
    int i = blockIdx.x * blockDim.x + threadIdx.x;
    if (i < NN * DD) h[i] = node_features[i];
    if (i < NN) deg[i] = 0;
}

__global__ __launch_bounds__(256) void count_deg_kernel(
    const int* __restrict__ src, int* __restrict__ deg)
{
    int i = blockIdx.x * blockDim.x + threadIdx.x;
    if (i < NE) atomicAdd(&deg[src[i]], 1);
}

// per-step: h -> bf16, agg -> 0
__global__ __launch_bounds__(256) void step_pre_kernel(
    const float* __restrict__ h, __bf16* __restrict__ hbf, float* __restrict__ agg)
{
    int i = blockIdx.x * blockDim.x + threadIdx.x;
    if (i < NN * DD) {
        hbf[i] = (__bf16)h[i];
        agg[i] = 0.0f;
    }
}

// ---------------- edge message kernel ----------------
// Each wave: 16 edges. msg[16x64] = [h_src | h_dst | e] (16x192 bf16) @ U^T (192x64 bf16)
// via 6 K-tiles x 4 N-tiles of v_wmma_f32_16x16x32_bf16. Weights staged in LDS once per
// block (served by ds_load_b128, bank-conflict-free); result scattered with f32 atomics
// into L2-resident agg.
__global__ __launch_bounds__(256) void edge_msg_kernel(
    const __bf16* __restrict__ hbf,       // [NN x 64]  bf16 (L2-resident, 6.4MB)
    const float*  __restrict__ edge_attr, // [NE x 64]  f32  (streamed, NT)
    const int*    __restrict__ src,
    const int*    __restrict__ dst,
    const __bf16* __restrict__ Uk,        // [64 x 192] bf16 (this step)
    const float*  __restrict__ Ub,        // [64] f32
    float*        __restrict__ agg)       // [NN x 64]  f32  (L2-resident, 12.8MB)
{
    __shared__ __bf16 Ush[DD * ULDS];     // 25.6 KB

    // cooperative stage of U into LDS: 64 rows x 24 16B-chunks = 1536 chunks
    for (int c = threadIdx.x; c < DD * (UKDIM / 8); c += 256) {
        const int row = c / (UKDIM / 8);
        const int off = (c % (UKDIM / 8)) * 8;
        *(uint4*)&Ush[row * ULDS + off] = *(const uint4*)(Uk + row * UKDIM + off);
    }
    __syncthreads();

    const int lane = threadIdx.x & 31;
    const int wave = threadIdx.x >> 5;
    const int tile = blockIdx.x * 8 + wave;   // 16 edges per wave-tile (exact: NE/16)
    const int e0   = tile * 16;
    const int m    = lane & 15;
    const int hi   = lane >> 4;

    const int e      = e0 + m;
    const int s_row  = src[e];
    const int d_row  = dst[e];
    const __bf16* hs = hbf + (size_t)s_row * DD;
    const __bf16* hd = hbf + (size_t)d_row * DD;
    const float*  ea = edge_attr + (size_t)e * DD;

    v8f acc[4];
#pragma unroll
    for (int n = 0; n < 4; ++n) { v8f z = {}; acc[n] = z; }

#pragma unroll
    for (int t = 0; t < 6; ++t) {
        const int kb0 = t * 32 + hi * 8;   // lanes 0-15: K=kb0..+7 & kb0+16..+23 ; lanes 16-31: +8
        const int kb1 = kb0 + 16;
        ABfrag a;
        if (t < 2) {                       // K in [0,64): h[src] gather (L2 hit)
            a.q[0] = *(const uint4*)(hs + kb0);
            a.q[1] = *(const uint4*)(hs + kb1);
        } else if (t < 4) {                // K in [64,128): h[dst] gather (L2 hit)
            a.q[0] = *(const uint4*)(hd + (kb0 - 64));
            a.q[1] = *(const uint4*)(hd + (kb1 - 64));
        } else {                           // K in [128,192): edge_attr stream, NT, f32->bf16
            cvt8_f32_bf16_nt(ea + (kb0 - 128), a.h);
            cvt8_f32_bf16_nt(ea + (kb1 - 128), a.h + 8);
        }
#pragma unroll
        for (int n = 0; n < 4; ++n) {
            const __bf16* Urow = &Ush[((lane & 15) + n * 16) * ULDS];
            ABfrag b;
            b.q[0] = *(const uint4*)(Urow + kb0);   // ds_load_b128, conflict-free
            b.q[1] = *(const uint4*)(Urow + kb1);
            acc[n] = __builtin_amdgcn_wmma_f32_16x16x32_bf16(
                false, a.v, false, b.v, (short)0, acc[n], false, false);
        }
    }

    // scatter-add: lane holds D[row = j + 8*hi][col = (lane&15) + 16*n] in acc[n][j]
    int sv[8];
#pragma unroll
    for (int j = 0; j < 8; ++j) sv[j] = src[e0 + hi * 8 + j];

#pragma unroll
    for (int n = 0; n < 4; ++n) {
        const int col = (lane & 15) + n * 16;
        const float bias = Ub[col];
#pragma unroll
        for (int j = 0; j < 8; ++j) {
            atomicAdd(&agg[(size_t)sv[j] * DD + col], acc[n][j] + bias);
        }
    }
}

// ---------------- node update kernel ----------------
// Each wave: 16 nodes. upd = relu([h | agg] (16x128 bf16) @ M^T (128x64)); masked in-place write.
__global__ __launch_bounds__(256) void node_update_kernel(
    const __bf16* __restrict__ hbf,   // [NN x 64] bf16
    const float*  __restrict__ agg,   // [NN x 64] f32
    const __bf16* __restrict__ Mk,    // [64 x 128] bf16 (this step)
    const float*  __restrict__ Mb,    // [64]
    const int*    __restrict__ deg,
    float*        __restrict__ h)     // [NN x 64] f32, updated in place
{
    __shared__ __bf16 Msh[DD * MLDS];     // 17.4 KB

    // stage M before any wave can exit (barrier must be block-wide)
    for (int c = threadIdx.x; c < DD * (MKDIM / 8); c += 256) {
        const int row = c / (MKDIM / 8);
        const int off = (c % (MKDIM / 8)) * 8;
        *(uint4*)&Msh[row * MLDS + off] = *(const uint4*)(Mk + row * MKDIM + off);
    }
    __syncthreads();

    const int lane = threadIdx.x & 31;
    const int wave = threadIdx.x >> 5;
    const int tile = blockIdx.x * 8 + wave;
    if (tile < NN / 16) {                 // wave-uniform guard, EXEC all-ones inside
        const int v0 = tile * 16;
        const int m  = lane & 15;
        const int hi = lane >> 4;

        const int node   = v0 + m;
        const __bf16* hr = hbf + (size_t)node * DD;
        const float*  ar = agg + (size_t)node * DD;

        v8f acc[4];
#pragma unroll
        for (int n = 0; n < 4; ++n) { v8f z = {}; acc[n] = z; }

#pragma unroll
        for (int t = 0; t < 4; ++t) {
            const int kb0 = t * 32 + hi * 8;
            const int kb1 = kb0 + 16;
            ABfrag a;
            if (t < 2) {                      // K in [0,64): h
                a.q[0] = *(const uint4*)(hr + kb0);
                a.q[1] = *(const uint4*)(hr + kb1);
            } else {                          // K in [64,128): agg (f32 -> bf16)
                cvt8_f32_bf16(ar + (kb0 - 64), a.h);
                cvt8_f32_bf16(ar + (kb1 - 64), a.h + 8);
            }
#pragma unroll
            for (int n = 0; n < 4; ++n) {
                const __bf16* Mrow = &Msh[((lane & 15) + n * 16) * MLDS];
                ABfrag b;
                b.q[0] = *(const uint4*)(Mrow + kb0);
                b.q[1] = *(const uint4*)(Mrow + kb1);
                acc[n] = __builtin_amdgcn_wmma_f32_16x16x32_bf16(
                    false, a.v, false, b.v, (short)0, acc[n], false, false);
            }
        }

#pragma unroll
        for (int n = 0; n < 4; ++n) {
            const int col = (lane & 15) + n * 16;
            const float bias = Mb[col];
#pragma unroll
            for (int j = 0; j < 8; ++j) {
                const int row  = j + hi * 8;
                const int nd   = v0 + row;
                float u = acc[n][j] + bias;
                u = fmaxf(u, 0.0f);
                if (deg[nd] > 0) h[(size_t)nd * DD + col] = u;   // else keep old h
            }
        }
    }
}

// ---------------- host launcher ----------------
extern "C" void kernel_launch(void* const* d_in, const int* in_sizes, int n_in,
                              void* d_out, int out_size, void* d_ws, size_t ws_size,
                              hipStream_t stream) {
    const float* node_features = (const float*)d_in[0];
    const float* edge_attr     = (const float*)d_in[1];
    const int*   src           = (const int*)d_in[2];
    const int*   dst           = (const int*)d_in[3];
    const float* U_w           = (const float*)d_in[4];
    const float* U_b           = (const float*)d_in[5];
    const float* M_w           = (const float*)d_in[6];
    const float* M_b           = (const float*)d_in[7];
    float* out = (float*)d_out;

    // workspace layout (all 256B aligned)
    auto al = [](size_t x) { return (x + 255) & ~(size_t)255; };
    char* p = (char*)d_ws;
    float*  h    = (float*)p;  p += al((size_t)NN * DD * sizeof(float));
    float*  agg  = (float*)p;  p += al((size_t)NN * DD * sizeof(float));
    __bf16* hbf  = (__bf16*)p; p += al((size_t)NN * DD * sizeof(__bf16));
    int*    deg  = (int*)p;    p += al((size_t)NN * sizeof(int));
    __bf16* Ubf  = (__bf16*)p; p += al((size_t)TSTEPS * DD * UKDIM * sizeof(__bf16));
    __bf16* Mbf  = (__bf16*)p; p += al((size_t)TSTEPS * DD * MKDIM * sizeof(__bf16));
    (void)ws_size; (void)in_sizes; (void)n_in; (void)out_size;

    // one-time prep
    prep_weights_kernel<<<(TSTEPS * DD * UKDIM + 255) / 256, 256, 0, stream>>>(U_w, M_w, Ubf, Mbf);
    init_kernel<<<(NN * DD + 255) / 256, 256, 0, stream>>>(node_features, h, deg);
    count_deg_kernel<<<(NE + 255) / 256, 256, 0, stream>>>(src, deg);

    // output[0 : NE*64] = edge_attr (passthrough)
    (void)hipMemcpyAsync(out, edge_attr, (size_t)NE * DD * sizeof(float),
                         hipMemcpyDeviceToDevice, stream);

    for (int k = 0; k < TSTEPS; ++k) {
        step_pre_kernel<<<(NN * DD + 255) / 256, 256, 0, stream>>>(h, hbf, agg);
        edge_msg_kernel<<<NE / 128, 256, 0, stream>>>(
            hbf, edge_attr, src, dst,
            Ubf + (size_t)k * DD * UKDIM, U_b + (size_t)k * DD, agg);
        node_update_kernel<<<(NN / 16 + 7) / 8, 256, 0, stream>>>(
            hbf, agg,
            Mbf + (size_t)k * DD * MKDIM, M_b + (size_t)k * DD, deg, h);
    }

    // output[NE*64 : NE*64 + NN*64] = final h
    (void)hipMemcpyAsync(out + (size_t)NE * DD, h, (size_t)NN * DD * sizeof(float),
                         hipMemcpyDeviceToDevice, stream);
}
